// MoE_9423158247593
// MI455X (gfx1250) — compile-verified, matrix operand order checked by reading, not run on hardware
//
#include <hip/hip_runtime.h>
#include <hip/hip_bf16.h>
#include <math.h>
#include <stdint.h>

typedef __bf16 bf16;
typedef __attribute__((ext_vector_type(16))) __bf16 v16bf;
typedef __attribute__((ext_vector_type(4)))  __bf16 v4bf;
typedef __attribute__((ext_vector_type(8)))  float  v8f;

#define N_TOK 16384
#define DDIM  1024
#define HDIM  4096
#define ODIM  1024
#define NEXP  8
#define NBAND 4
#define RRANK 16
#define SCALE_LORA 2.0f
#define CAP_SLOTS (2 * N_TOK + NEXP * 128)   // 33792, multiple of 8 and 128

union FragU { v16bf v; uint4 q[2]; };

// ---- CDNA5 async global->LDS copy (ASYNCcnt path, no VGPR round-trip) ----
// lds_off is the workgroup-relative LDS byte offset (low 32 bits of a generic
// LDS pointer, ISA 10.2).
__device__ __forceinline__ void async_copy_b128(const bf16* gsrc, unsigned lds_off) {
  unsigned long long ga = (unsigned long long)(uintptr_t)gsrc;
  asm volatile("global_load_async_to_lds_b128 %0, %1, off"
               :: "v"(lds_off), "v"(ga)
               : "memory");
}
__device__ __forceinline__ void wait_asynccnt0() {
  asm volatile("s_wait_asynccnt 0x0" ::: "memory");
}

// ---------------------------------------------------------------- fp32 -> bf16 (layout preserved)
__global__ void cvt_kernel(const float* __restrict__ s, bf16* __restrict__ d, size_t n4) {
  size_t i = (size_t)blockIdx.x * blockDim.x + threadIdx.x;
  size_t stride = (size_t)gridDim.x * blockDim.x;
  for (; i < n4; i += stride) {
    float4 f = ((const float4*)s)[i];
    v4bf o;
    o[0] = (bf16)f.x; o[1] = (bf16)f.y; o[2] = (bf16)f.z; o[3] = (bf16)f.w;
    ((v4bf*)d)[i] = o;
  }
}

// ---------------------------------------------------------------- fp32 [E][K][N] -> bf16 [E][N][K]
__global__ __launch_bounds__(256) void cvt_transpose_kernel(
    const float* __restrict__ W, bf16* __restrict__ WT, int Kdim, int Ncols) {
  __shared__ bf16 tile[32][40];
  int e  = blockIdx.z;
  int n0 = blockIdx.x * 32;
  int k0 = blockIdx.y * 32;
  const float* Wp = W + ((size_t)e * Kdim + k0) * Ncols + n0;
  bf16* WTp = WT + ((size_t)e * Ncols + n0) * Kdim + k0;
  int t  = threadIdx.x;
  int r  = t >> 3;         // 0..31
  int c4 = (t & 7) * 4;    // 0..28
  float4 f = *(const float4*)(Wp + (size_t)r * Ncols + c4);
  tile[r][c4 + 0] = (bf16)f.x;
  tile[r][c4 + 1] = (bf16)f.y;
  tile[r][c4 + 2] = (bf16)f.z;
  tile[r][c4 + 3] = (bf16)f.w;
  __syncthreads();
  v4bf o;
  o[0] = tile[c4 + 0][r];
  o[1] = tile[c4 + 1][r];
  o[2] = tile[c4 + 2][r];
  o[3] = tile[c4 + 3][r];
  *(v4bf*)(WTp + (size_t)r * Kdim + c4) = o;
}

// ---------------------------------------------------------------- gating
__global__ __launch_bounds__(256) void gating_kernel(
    const float* __restrict__ x, const float* __restrict__ w_gate,
    int* __restrict__ tok_e, float* __restrict__ tok_w, float* __restrict__ partials) {
  __shared__ float wg[DDIM * NEXP];          // 32 KB
  __shared__ float wimp[8][NEXP];
  __shared__ float wcnt[8][NEXP];
  int tid = threadIdx.x;
  for (int i = tid; i < DDIM * NEXP; i += 256) wg[i] = w_gate[i];
  __syncthreads();
  int wave = tid >> 5, lane = tid & 31;
  float imp[NEXP], cnt[NEXP];
#pragma unroll
  for (int e = 0; e < NEXP; ++e) { imp[e] = 0.f; cnt[e] = 0.f; }
  int tbase = blockIdx.x * 256 + wave * 32;
  for (int it = 0; it < 32; ++it) {
    int t = tbase + it;
    const float* xr = x + (size_t)t * DDIM;
    float acc[NEXP];
#pragma unroll
    for (int e = 0; e < NEXP; ++e) acc[e] = 0.f;
    for (int d = lane; d < DDIM; d += 32) {
      float xv = xr[d];
#pragma unroll
      for (int e = 0; e < NEXP; ++e) acc[e] = fmaf(xv, wg[d * NEXP + e], acc[e]);
    }
#pragma unroll
    for (int e = 0; e < NEXP; ++e)
      for (int off = 16; off; off >>= 1) acc[e] += __shfl_down(acc[e], off, 32);
    if (lane == 0) {
      int b1i = 0; float b1v = acc[0];
      for (int e = 1; e < NEXP; ++e) if (acc[e] > b1v) { b1v = acc[e]; b1i = e; }
      int b2i = -1; float b2v = -1e30f;
      for (int e = 0; e < NEXP; ++e) if (e != b1i && acc[e] > b2v) { b2v = acc[e]; b2i = e; }
      float z  = expf(b2v - b1v);
      float w1 = 1.0f / (1.0f + z);
      float w2 = z / (1.0f + z);
      tok_e[2 * t] = b1i; tok_e[2 * t + 1] = b2i;
      tok_w[2 * t] = w1;  tok_w[2 * t + 1] = w2;
      imp[b1i] += w1; imp[b2i] += w2;
      cnt[b1i] += 1.f; cnt[b2i] += 1.f;
    }
  }
  if (lane == 0) {
#pragma unroll
    for (int e = 0; e < NEXP; ++e) { wimp[wave][e] = imp[e]; wcnt[wave][e] = cnt[e]; }
  }
  __syncthreads();
  if (tid < NEXP) {
    float si = 0.f, sc = 0.f;
    for (int w = 0; w < 8; ++w) { si += wimp[w][tid]; sc += wcnt[w][tid]; }
    partials[blockIdx.x * 16 + tid]     = si;
    partials[blockIdx.x * 16 + 8 + tid] = sc;
  }
}

// ---------------------------------------------------------------- finalize: loss + segment offsets
__device__ float cv2_dev(const float* v) {
  float m = 0.f;
  for (int i = 0; i < NEXP; ++i) m += v[i];
  m *= (1.0f / NEXP);
  float var = 0.f;
  for (int i = 0; i < NEXP; ++i) { float d = v[i] - m; var += d * d; }
  var *= (1.0f / (NEXP - 1));
  return var / (m * m + 1e-10f);
}

__global__ void finalize_kernel(const float* __restrict__ partials, int* __restrict__ counts,
                                int* __restrict__ offsets, int* __restrict__ cursors,
                                float* __restrict__ loss_out) {
  if (threadIdx.x != 0 || blockIdx.x != 0) return;
  float imp[NEXP], ld[NEXP];
  for (int e = 0; e < NEXP; ++e) { imp[e] = 0.f; ld[e] = 0.f; }
  for (int b = 0; b < 64; ++b)
    for (int e = 0; e < NEXP; ++e) {
      imp[e] += partials[b * 16 + e];
      ld[e]  += partials[b * 16 + 8 + e];
    }
  loss_out[0] = 0.01f * (cv2_dev(imp) + cv2_dev(ld));
  int off = 0;
  for (int e = 0; e < NEXP; ++e) {
    int c = (int)(ld[e] + 0.5f);
    counts[e] = c;
    offsets[e] = off;
    cursors[e] = 0;
    off += (c + 127) & ~127;
  }
}

// ---------------------------------------------------------------- scatter tokens into expert segments
__global__ void scatter_kernel(const int* __restrict__ tok_e, const float* __restrict__ tok_w,
                               const int* __restrict__ offsets, int* __restrict__ cursors,
                               int* __restrict__ slot_token, float* __restrict__ slot_gate,
                               int* __restrict__ slot_expert) {
  int t = blockIdx.x * blockDim.x + threadIdx.x;
  if (t >= N_TOK) return;
#pragma unroll
  for (int k = 0; k < 2; ++k) {
    int e  = tok_e[2 * t + k];
    float w = tok_w[2 * t + k];
    int pos = atomicAdd(&cursors[e], 1);
    int s = offsets[e] + pos;
    slot_token[s]  = t;
    slot_gate[s]   = w;
    slot_expert[s] = e;
  }
}

// ---------------------------------------------------------------- rank-16 LoRA projection
template <int KD, int SRCBF>
__global__ __launch_bounds__(256) void lora_t_kernel(
    const float* __restrict__ xsrc, const bf16* __restrict__ hsrc,
    const float* __restrict__ Amat, const int* __restrict__ slot_token,
    const int* __restrict__ slot_expert, const int* __restrict__ band_idx,
    float* __restrict__ tout) {
  int wave = threadIdx.x >> 5, lane = threadIdx.x & 31;
  int slot = blockIdx.x * 8 + wave;
  if (slot >= CAP_SLOTS) return;
  int tok = slot_token[slot];
  if (tok < 0) return;
  int e = slot_expert[slot];
  int band = band_idx[tok];
  const float* Ab = Amat + ((size_t)e * NBAND + band) * (size_t)KD * RRANK;
  float acc[RRANK];
#pragma unroll
  for (int r = 0; r < RRANK; ++r) acc[r] = 0.f;
  if (SRCBF) {
    const bf16* hr = hsrc + (size_t)slot * KD;
    for (int d = lane; d < KD; d += 32) {
      float hv = (float)hr[d];
      const float* ar = Ab + (size_t)d * RRANK;
#pragma unroll
      for (int r = 0; r < RRANK; ++r) acc[r] = fmaf(hv, ar[r], acc[r]);
    }
  } else {
    const float* xr = xsrc + (size_t)tok * KD;
    for (int d = lane; d < KD; d += 32) {
      float xv = xr[d];
      const float* ar = Ab + (size_t)d * RRANK;
#pragma unroll
      for (int r = 0; r < RRANK; ++r) acc[r] = fmaf(xv, ar[r], acc[r]);
    }
  }
#pragma unroll
  for (int r = 0; r < RRANK; ++r)
    for (int off = 16; off; off >>= 1) acc[r] += __shfl_down(acc[r], off, 32);
  if (lane == 0) {
    float* tp = tout + (size_t)slot * RRANK;
#pragma unroll
    for (int r = 0; r < RRANK; ++r) tp[r] = acc[r];
  }
}

// ---------------------------------------------------------------- grouped bf16 WMMA GEMM
// Tile: 128(M) x 128(N) x 64(K); 8 waves as 4(M) x 2(N); 16 WMMAs per stage.
// Double-buffered LDS stage filled by CDNA5 global_load_async_to_lds_b128:
// iteration top waits ASYNCcnt for buffer `cur`, one barrier, then issues the
// asyncs for buffer `next` while WMMAs consume `cur` (copy/compute overlap).
// Padding rows stream from a zeroed page (branchless staging).
template <int KDIM, int NCOLS, int EPI>
__global__ __launch_bounds__(256) void gemm_kernel(
    const bf16* __restrict__ A, const bf16* __restrict__ BwT,
    const float* __restrict__ biasv, const float* __restrict__ loraB,
    const float* __restrict__ tvec, const int* __restrict__ slot_token,
    const float* __restrict__ slot_gate, const int* __restrict__ band_idx,
    const int* __restrict__ counts, const int* __restrict__ offsets,
    const bf16* __restrict__ zpage,
    bf16* __restrict__ hb_out, float* __restrict__ y_out) {
  struct Stage { bf16 a[128][72]; bf16 b[128][72]; };     // 36864 B
  __shared__ union SM {
    Stage stage[2];                                        // 73728 B
    float lb[NBAND][RRANK][128];                           // 32 KB lora-B slice
  } sm;
  __shared__ int rowTok[128];
  __shared__ int rowBand[128];

  const int e = blockIdx.y;
  const int cnt = counts[e];
  const int m0 = blockIdx.x * 128;
  if (m0 >= cnt) return;
  const int seg = offsets[e];
  const int n0 = blockIdx.z * 128;
  const int tid = threadIdx.x;

  if (tid < 128) {
    int t = slot_token[seg + m0 + tid];
    rowTok[tid] = t;
    rowBand[tid] = (t >= 0) ? band_idx[t] : 0;
  }
  __syncthreads();

  const int wave = tid >> 5, lane = tid & 31;
  const int wm = wave & 3, wn = wave >> 2;   // 4 x 2 wave grid
  const int lr = lane & 15;
  const int koff = (lane < 16) ? 0 : 8;
  const unsigned stageSz = (unsigned)sizeof(Stage);

  // branchless staging state: per-chunk source pointer + step (0 for padding rows)
  const bf16* Bp = BwT + ((size_t)e * NCOLS + n0) * KDIM;
  const bf16* aPtr[4];
  const bf16* bPtr[4];
  int aStep[4];
  unsigned aOff[4], bOff[4];                 // LDS offsets within stage[0]
#pragma unroll
  for (int p = 0; p < 4; ++p) {
    int c = tid + p * 256;
    int r = c >> 3;            // 0..127
    int cc = (c & 7) * 8;      // 0..56
    int t = rowTok[r];
    if (EPI == 1) {
      bool v = (t >= 0);
      aPtr[p]  = v ? (A + (size_t)t * KDIM + cc) : zpage;
      aStep[p] = v ? 64 : 0;
    } else {
      aPtr[p]  = A + (size_t)(seg + m0 + r) * KDIM + cc;
      aStep[p] = 64;
    }
    aOff[p] = (unsigned)(uintptr_t)&sm.stage[0].a[r][cc];
    bPtr[p] = Bp + (size_t)r * KDIM + cc;
    bOff[p] = (unsigned)(uintptr_t)&sm.stage[0].b[r][cc];
  }

  v8f zero = {0.f, 0.f, 0.f, 0.f, 0.f, 0.f, 0.f, 0.f};
  v8f acc[2][4];
#pragma unroll
  for (int mt = 0; mt < 2; ++mt)
#pragma unroll
    for (int nt = 0; nt < 4; ++nt) acc[mt][nt] = zero;

  // prologue: fill buffer 0
#pragma unroll
  for (int p = 0; p < 4; ++p) {
    async_copy_b128(aPtr[p], aOff[p]);
    aPtr[p] += aStep[p];
    async_copy_b128(bPtr[p], bOff[p]);
    bPtr[p] += 64;
  }

  unsigned buf = 0;
  for (int k0 = 0; k0 < KDIM; k0 += 64) {
    wait_asynccnt0();
    __syncthreads();

    // issue next stage into the other buffer while computing on this one
    if (k0 + 64 < KDIM) {
      unsigned nb = (buf ^ 1u) * stageSz;
#pragma unroll
      for (int p = 0; p < 4; ++p) {
        async_copy_b128(aPtr[p], aOff[p] + nb);
        aPtr[p] += aStep[p];
        async_copy_b128(bPtr[p], bOff[p] + nb);
        bPtr[p] += 64;
        __builtin_prefetch((const void*)(bPtr[p]), 0, 0);
      }
    }

    const Stage* st = &sm.stage[buf];
    // --- fragments + WMMA: two K=32 sub-steps
#pragma unroll
    for (int ks = 0; ks < 2; ++ks) {
      const int kb = ks * 32 + koff;
      v16bf afrag[2], bfrag[4];
#pragma unroll
      for (int mt = 0; mt < 2; ++mt) {
        FragU u;
        const bf16* ap = &st->a[wm * 32 + mt * 16 + lr][kb];
        u.q[0] = *(const uint4*)(ap);
        u.q[1] = *(const uint4*)(ap + 16);
        afrag[mt] = u.v;
      }
#pragma unroll
      for (int nt = 0; nt < 4; ++nt) {
        FragU u;
        const bf16* bp = &st->b[wn * 64 + nt * 16 + lr][kb];
        u.q[0] = *(const uint4*)(bp);
        u.q[1] = *(const uint4*)(bp + 16);
        bfrag[nt] = u.v;
      }
#pragma unroll
      for (int mt = 0; mt < 2; ++mt)
#pragma unroll
        for (int nt = 0; nt < 4; ++nt)
          acc[mt][nt] = __builtin_amdgcn_wmma_f32_16x16x32_bf16(
              false, afrag[mt], false, bfrag[nt], (short)0, acc[mt][nt], false, false);
    }
    buf ^= 1u;
  }
  __syncthreads();

  // --- epilogue: stage lora-B slices for all 4 bands
  for (int i = tid; i < NBAND * RRANK * 128; i += 256) {
    int b  = i >> 11;
    int rr = (i >> 7) & 15;
    int nn = i & 127;
    sm.lb[b][rr][nn] =
        loraB[(((size_t)e * NBAND + b) * RRANK + rr) * NCOLS + n0 + nn];
  }
  __syncthreads();

#pragma unroll
  for (int mt = 0; mt < 2; ++mt) {
#pragma unroll
    for (int i = 0; i < 8; ++i) {
      int row = wm * 32 + mt * 16 + i + ((lane < 16) ? 0 : 8);
      int tok = rowTok[row];
      int band = rowBand[row];
      int slotAbs = seg + m0 + row;
      float tv[RRANK];
#pragma unroll
      for (int r = 0; r < RRANK; ++r) tv[r] = 0.f;
      if (tok >= 0) {
        const float* tp = tvec + (size_t)slotAbs * RRANK;
#pragma unroll
        for (int r = 0; r < RRANK; ++r) tv[r] = tp[r];
      }
      float gate = 0.f;
      if (EPI == 2 && tok >= 0) gate = slot_gate[slotAbs];
#pragma unroll
      for (int nt = 0; nt < 4; ++nt) {
        int ncol = wn * 64 + nt * 16 + lr;
        float v = acc[mt][nt][i];
        v += biasv[(size_t)e * NCOLS + n0 + ncol];
        float lora = 0.f;
#pragma unroll
        for (int r = 0; r < RRANK; ++r) lora = fmaf(tv[r], sm.lb[band][r][ncol], lora);
        v += SCALE_LORA * lora;
        if (EPI == 1) {
          float g = 0.5f * v * (1.0f + erff(v * 0.70710678118f));
          hb_out[(size_t)slotAbs * NCOLS + n0 + ncol] = (tok >= 0) ? (bf16)g : (bf16)0.f;
        } else {
          if (tok >= 0)
            atomicAdd(&y_out[(size_t)tok * NCOLS + n0 + ncol], gate * v);
        }
      }
    }
  }
}

// ---------------------------------------------------------------- launch
extern "C" void kernel_launch(void* const* d_in, const int* in_sizes, int n_in,
                              void* d_out, int out_size, void* d_ws, size_t ws_size,
                              hipStream_t stream) {
  const float* x      = (const float*)d_in[0];
  const int*   band   = (const int*)d_in[1];
  const float* w_gate = (const float*)d_in[2];
  const float* W1     = (const float*)d_in[3];
  const float* b1     = (const float*)d_in[4];
  const float* W2     = (const float*)d_in[5];
  const float* b2     = (const float*)d_in[6];
  const float* A1     = (const float*)d_in[7];
  const float* B1     = (const float*)d_in[8];
  const float* A2     = (const float*)d_in[9];
  const float* B2     = (const float*)d_in[10];
  float* y = (float*)d_out;
  float* loss = y + (size_t)N_TOK * ODIM;

  char* base = (char*)d_ws;
  size_t off = 0;
  auto carve = [&](size_t bytes) {
    size_t o = off;
    off = (off + bytes + 255) & ~(size_t)255;
    return o;
  };
  size_t o_xb   = carve((size_t)N_TOK * DDIM * 2);
  size_t o_w1b  = carve((size_t)NEXP * DDIM * HDIM * 2);   // transposed [E][H][D]
  size_t o_w2b  = carve((size_t)NEXP * HDIM * ODIM * 2);   // transposed [E][O][H]
  size_t o_hb   = carve((size_t)CAP_SLOTS * HDIM * 2);
  size_t o_t1   = carve((size_t)CAP_SLOTS * RRANK * 4);
  size_t o_t2   = carve((size_t)CAP_SLOTS * RRANK * 4);
  size_t o_te   = carve((size_t)N_TOK * 2 * 4);
  size_t o_tw   = carve((size_t)N_TOK * 2 * 4);
  size_t o_st   = carve((size_t)CAP_SLOTS * 4);
  size_t o_sg   = carve((size_t)CAP_SLOTS * 4);
  size_t o_se   = carve((size_t)CAP_SLOTS * 4);
  size_t o_part = carve((size_t)64 * 16 * 4);
  size_t o_cnt  = carve((size_t)NEXP * 4);
  size_t o_off  = carve((size_t)NEXP * 4);
  size_t o_cur  = carve((size_t)NEXP * 4);
  size_t o_zp   = carve((size_t)256);                       // zero page
  (void)ws_size; (void)in_sizes; (void)n_in; (void)out_size;

  bf16*  xb   = (bf16*)(base + o_xb);
  bf16*  W1bT = (bf16*)(base + o_w1b);
  bf16*  W2bT = (bf16*)(base + o_w2b);
  bf16*  hb   = (bf16*)(base + o_hb);
  float* t1   = (float*)(base + o_t1);
  float* t2   = (float*)(base + o_t2);
  int*   tokE = (int*)(base + o_te);
  float* tokW = (float*)(base + o_tw);
  int*   sTok = (int*)(base + o_st);
  float* sGat = (float*)(base + o_sg);
  int*   sExp = (int*)(base + o_se);
  float* part = (float*)(base + o_part);
  int*   cnts = (int*)(base + o_cnt);
  int*   offs = (int*)(base + o_off);
  int*   curs = (int*)(base + o_cur);
  bf16*  zp   = (bf16*)(base + o_zp);

  hipMemsetAsync(y, 0, (size_t)N_TOK * ODIM * sizeof(float), stream);
  hipMemsetAsync(sTok, 0xFF, (size_t)CAP_SLOTS * 4, stream);   // token = -1
  hipMemsetAsync(zp, 0, 256, stream);

  cvt_kernel<<<2048, 256, 0, stream>>>(x, xb, (size_t)N_TOK * DDIM / 4);
  cvt_transpose_kernel<<<dim3(HDIM / 32, DDIM / 32, NEXP), 256, 0, stream>>>(W1, W1bT, DDIM, HDIM);
  cvt_transpose_kernel<<<dim3(ODIM / 32, HDIM / 32, NEXP), 256, 0, stream>>>(W2, W2bT, HDIM, ODIM);

  gating_kernel<<<64, 256, 0, stream>>>(x, w_gate, tokE, tokW, part);
  finalize_kernel<<<1, 32, 0, stream>>>(part, cnts, offs, curs, loss);
  scatter_kernel<<<64, 256, 0, stream>>>(tokE, tokW, offs, curs, sTok, sGat, sExp);

  lora_t_kernel<DDIM, 0><<<CAP_SLOTS / 8, 256, 0, stream>>>(
      x, (const bf16*)nullptr, A1, sTok, sExp, band, t1);

  gemm_kernel<DDIM, HDIM, 1><<<dim3(128, NEXP, HDIM / 128), 256, 0, stream>>>(
      xb, W1bT, b1, B1, t1, sTok, sGat, band, cnts, offs, zp, hb, nullptr);

  lora_t_kernel<HDIM, 1><<<CAP_SLOTS / 8, 256, 0, stream>>>(
      (const float*)nullptr, hb, A2, sTok, sExp, band, t2);

  gemm_kernel<HDIM, ODIM, 2><<<dim3(128, NEXP, ODIM / 128), 256, 0, stream>>>(
      hb, W2bT, b2, B2, t2, sTok, sGat, band, cnts, offs, zp, nullptr, y);
}
